// GAT_with_global_update_83468394431131
// MI455X (gfx1250) — matrix operand order repaired and checked by reading, not compiled, not run on hardware
//
#include <hip/hip_runtime.h>
#include <hip/hip_bf16.h>

#define NN   50000
#define EE   400000
#define DD   128
#define HH   128
#define GG   8
#define NHOPS 3
#define GOUTD 128

typedef __attribute__((ext_vector_type(16))) __bf16 v16bf;
typedef __attribute__((ext_vector_type(8)))  float  v8f;

static __host__ int cdiv(int a, int b) { return (a + b - 1) / b; }

// ---------------------------------------------------------------------------
// q = X @ W + bias, X:[nrows,128] f32, W:[128,128] f32, Q:[nrows,128] f32
// One wave computes a 16x128 output stripe with v_wmma_f32_16x16x32_bf16.
// ---------------------------------------------------------------------------
__global__ __launch_bounds__(128) void gemm_q_bf16(
    const float* __restrict__ X, const float* __restrict__ W,
    const float* __restrict__ bias, float* __restrict__ Q, int nrows)
{
    const int wave = blockIdx.x * (blockDim.x >> 5) + (threadIdx.x >> 5);
    const int lane = threadIdx.x & 31;
    const int row0 = wave * 16;
    if (row0 >= nrows) return;

    // A fragment mapping (16x32 bf16): lanes 0-15 row M=lane, K 0..15;
    // lanes 16-31 row M=lane-16, K 16..31.
    const int a_m    = lane & 15;
    const int a_koff = (lane >> 4) << 4;   // 0 or 16
    // B fragment mapping (32x16 bf16): lanes 0-15 col N=lane, K 0..15;
    // lanes 16-31 col N=lane-16, K 16..31.
    const int b_n    = lane & 15;
    const int b_koff = (lane >> 4) << 4;   // 0 or 16

    v8f acc[8] = {};
    const float* xrow = X + (size_t)(row0 + a_m) * DD;

    for (int k0 = 0; k0 < DD; k0 += 32) {
        v16bf a;
#pragma unroll
        for (int i = 0; i < 8; ++i) a[i]     = (__bf16)xrow[k0 + a_koff + i];
#pragma unroll
        for (int i = 0; i < 8; ++i) a[8 + i] = (__bf16)xrow[k0 + 8 + a_koff + i];

#pragma unroll
        for (int nt = 0; nt < 8; ++nt) {
            v16bf b;
            const float* wp = W + (size_t)(k0 + b_koff) * HH + nt * 16 + b_n;
#pragma unroll
            for (int j = 0; j < 16; ++j) b[j] = (__bf16)wp[j * HH];
            acc[nt] = __builtin_amdgcn_wmma_f32_16x16x32_bf16(
                false, a, false, b, (short)0, acc[nt], false, false);
        }
    }

    // C layout: VGPR r -> M = 8*(lane>=16) + r, N = lane&15 within tile.
    const int c_n  = lane & 15;
    const int c_m0 = (lane >> 4) << 3;
#pragma unroll
    for (int nt = 0; nt < 8; ++nt) {
#pragma unroll
        for (int r = 0; r < 8; ++r) {
            const int col = nt * 16 + c_n;
            Q[(size_t)(row0 + c_m0 + r) * HH + col] = acc[nt][r] + bias[col];
        }
    }
}

// ---------------------------------------------------------------------------
// a_s[n] = q[n] . Wa[0:128],  a_r[n] = q[n] . Wa[128:256]  (one wave / node)
// ---------------------------------------------------------------------------
__global__ __launch_bounds__(256) void attn_coef(
    const float* __restrict__ Q, const float* __restrict__ Wa,
    float* __restrict__ a_s, float* __restrict__ a_r, int n)
{
    const int wave = blockIdx.x * (blockDim.x >> 5) + (threadIdx.x >> 5);
    const int lane = threadIdx.x & 31;
    if (wave >= n) return;
    const float* qr = Q + (size_t)wave * HH;
    float s0 = 0.f, s1 = 0.f;
#pragma unroll
    for (int i = lane; i < HH; i += 32) {
        const float qv = qr[i];
        s0 += qv * Wa[i];
        s1 += qv * Wa[HH + i];
    }
#pragma unroll
    for (int off = 16; off > 0; off >>= 1) {
        s0 += __shfl_xor(s0, off, 32);
        s1 += __shfl_xor(s1, off, 32);
    }
    if (lane == 0) { a_s[wave] = s0; a_r[wave] = s1; }
}

// zero x_next [N*H], segmax (ordered-uint -inf == 0), segsum
__global__ void hop_init(float* __restrict__ xnext, unsigned* __restrict__ segmax,
                         float* __restrict__ segsum, int nh, int n)
{
    const int i = blockIdx.x * blockDim.x + threadIdx.x;
    if (i < nh) xnext[i] = 0.f;
    if (i < n) { segmax[i] = 0u; segsum[i] = 0.f; }
}

__device__ __forceinline__ unsigned ord_encode(float f)
{
    unsigned u = __float_as_uint(f);
    return (u & 0x80000000u) ? ~u : (u | 0x80000000u);
}
__device__ __forceinline__ float ord_decode(unsigned u)
{
    return (u & 0x80000000u) ? __uint_as_float(u ^ 0x80000000u)
                             : __uint_as_float(~u);
}

__global__ void edge_max(const int* __restrict__ snd, const int* __restrict__ rcv,
                         const float* __restrict__ a_s, const float* __restrict__ a_r,
                         const float* __restrict__ ba, unsigned* __restrict__ segmax,
                         int e_cnt)
{
    const int e = blockIdx.x * blockDim.x + threadIdx.x;
    if (e >= e_cnt) return;
    const float logit = a_s[snd[e]] + a_r[rcv[e]] + ba[0];
    atomicMax(&segmax[rcv[e]], ord_encode(logit));
}

__global__ void edge_sum(const int* __restrict__ snd, const int* __restrict__ rcv,
                         const float* __restrict__ a_s, const float* __restrict__ a_r,
                         const float* __restrict__ ba, const unsigned* __restrict__ segmax,
                         float* __restrict__ ew, float* __restrict__ segsum, int e_cnt)
{
    const int e = blockIdx.x * blockDim.x + threadIdx.x;
    if (e >= e_cnt) return;
    const int r = rcv[e];
    const float logit = a_s[snd[e]] + a_r[r] + ba[0];
    const float v = expf(logit - ord_decode(segmax[r]));
    ew[e] = v;
    atomicAdd(&segsum[r], v);
}

// one wave per edge: x_next[rcv] += q[snd] * (ew / segsum[rcv])
__global__ __launch_bounds__(256) void edge_scatter(
    const int* __restrict__ snd, const int* __restrict__ rcv,
    const float* __restrict__ ew, const float* __restrict__ segsum,
    const float* __restrict__ Q, float* __restrict__ xnext, int e_cnt)
{
    const int wave = blockIdx.x * (blockDim.x >> 5) + (threadIdx.x >> 5);
    const int lane = threadIdx.x & 31;
    if (wave >= e_cnt) return;
    const int s = snd[wave], r = rcv[wave];
    const float w = ew[wave] / segsum[r];
    const float* qs = Q + (size_t)s * HH;
    float* xr = xnext + (size_t)r * HH;
#pragma unroll
    for (int i = lane; i < HH; i += 32) atomicAdd(&xr[i], qs[i] * w);
}

__global__ void leaky_relu_k(float* __restrict__ x, int n)
{
    const int i = blockIdx.x * blockDim.x + threadIdx.x;
    if (i >= n) return;
    const float v = x[i];
    x[i] = v > 0.f ? v : 0.01f * v;
}

__global__ void zero_f(float* __restrict__ p, int n)
{
    const int i = blockIdx.x * blockDim.x + threadIdx.x;
    if (i < n) p[i] = 0.f;
}

// node_graph_idx is sorted: per-thread register accumulation over a node chunk,
// flush one atomic per graph transition.  blockDim = 128 (one thread per dim).
#define AGG_CHUNK 256
__global__ __launch_bounds__(128) void agg_nodes(
    const float* __restrict__ x, const int* __restrict__ gidx,
    float* __restrict__ agg, int n)
{
    const int d  = threadIdx.x;
    const int n0 = blockIdx.x * AGG_CHUNK;
    if (n0 >= n) return;
    const int n1 = (n0 + AGG_CHUNK < n) ? n0 + AGG_CHUNK : n;
    int   cur  = gidx[n0];
    float accv = 0.f;
    for (int i = n0; i < n1; ++i) {
        const int g = gidx[i];
        if (g != cur) {
            atomicAdd(&agg[(size_t)cur * HH + d], accv);
            accv = 0.f; cur = g;
        }
        accv += x[(size_t)i * HH + d];
    }
    atomicAdd(&agg[(size_t)cur * HH + d], accv);
}

// g[gi,:] = concat(agg[gi,:], globals_[gi,:]) @ Wg + bg ; grid = G, block = GOUT
__global__ __launch_bounds__(128) void global_mlp(
    const float* __restrict__ agg, const float* __restrict__ globals_,
    const float* __restrict__ Wg, const float* __restrict__ bg,
    float* __restrict__ gout)
{
    const int j  = threadIdx.x;
    const int gi = blockIdx.x;
    float acc = bg[j];
    for (int k = 0; k < HH; ++k) acc += agg[(size_t)gi * HH + k] * Wg[(size_t)k * GOUTD + j];
    for (int k = 0; k < DD; ++k) acc += globals_[(size_t)gi * DD + k] * Wg[(size_t)(HH + k) * GOUTD + j];
    gout[(size_t)gi * GOUTD + j] = acc;
}

// ---------------------------------------------------------------------------
extern "C" void kernel_launch(void* const* d_in, const int* in_sizes, int n_in,
                              void* d_out, int out_size, void* d_ws, size_t ws_size,
                              hipStream_t stream)
{
    (void)in_sizes; (void)n_in; (void)out_size; (void)ws_size;

    const float* nodes    = (const float*)d_in[0];
    const float* globals_ = (const float*)d_in[1];
    const float* Wq       = (const float*)d_in[2];   // [3,128,128]
    const float* bq       = (const float*)d_in[3];   // [3,128]
    const float* Wa       = (const float*)d_in[4];   // [3,256]
    const float* ba       = (const float*)d_in[5];   // [3]
    const float* Wg       = (const float*)d_in[6];   // [256,128]
    const float* bg       = (const float*)d_in[7];   // [128]
    const int*   snd      = (const int*)d_in[8];
    const int*   rcv      = (const int*)d_in[9];
    const int*   gidx     = (const int*)d_in[10];

    float* out_x = (float*)d_out;                    // [N,128]
    float* out_g = out_x + (size_t)NN * HH;          // [G,128]

    // workspace layout
    float*    q      = (float*)d_ws;                 // N*H
    float*    xbuf   = q + (size_t)NN * HH;          // N*H
    float*    ew     = xbuf + (size_t)NN * HH;       // E
    unsigned* segmax = (unsigned*)(ew + EE);         // N
    float*    segsum = (float*)(segmax + NN);        // N
    float*    a_s    = segsum + NN;                  // N
    float*    a_r    = a_s + NN;                     // N
    float*    agg    = a_r + NN;                     // G*H

    const int gemm_waves = NN / 16;                              // 3125
    const dim3 gemm_grid(cdiv(gemm_waves, 4)), gemm_blk(128);
    const dim3 attn_grid(cdiv(NN, 8)), attn_blk(256);
    const dim3 init_grid(cdiv(NN * HH, 256)), b256(256);
    const dim3 edge_grid(cdiv(EE, 256));
    const dim3 scat_grid(cdiv(EE, 8));

    for (int h = 0; h < NHOPS; ++h) {
        const float* x_in  = (h == 0) ? nodes : xbuf;
        float*       x_out = (h == NHOPS - 1) ? out_x : xbuf;
        const float* Wq_h  = Wq + (size_t)h * DD * HH;
        const float* bq_h  = bq + (size_t)h * HH;
        const float* Wa_h  = Wa + (size_t)h * 2 * HH;
        const float* ba_h  = ba + h;

        // q = x_in @ Wq_h + bq_h   (stream order keeps x_in alive until done)
        gemm_q_bf16<<<gemm_grid, gemm_blk, 0, stream>>>(x_in, Wq_h, bq_h, q, NN);
        // per-node logit halves
        attn_coef<<<attn_grid, attn_blk, 0, stream>>>(q, Wa_h, a_s, a_r, NN);
        // zero x_out + softmax state (safe even when x_out == x_in: gemm done)
        hop_init<<<init_grid, b256, 0, stream>>>(x_out, segmax, segsum, NN * HH, NN);
        // segment softmax
        edge_max<<<edge_grid, b256, 0, stream>>>(snd, rcv, a_s, a_r, ba_h, segmax, EE);
        edge_sum<<<edge_grid, b256, 0, stream>>>(snd, rcv, a_s, a_r, ba_h, segmax, ew, segsum, EE);
        // weighted scatter-sum of q[snd] into x_out[rcv]
        edge_scatter<<<scat_grid, b256, 0, stream>>>(snd, rcv, ew, segsum, q, x_out, EE);
        // node update fn
        leaky_relu_k<<<init_grid, b256, 0, stream>>>(x_out, NN * HH);
    }

    // global update: segment-sum nodes per graph, then dense on concat
    zero_f<<<dim3(cdiv(GG * HH, 256)), b256, 0, stream>>>(agg, GG * HH);
    agg_nodes<<<dim3(cdiv(NN, AGG_CHUNK)), dim3(128), 0, stream>>>(out_x, gidx, agg, NN);
    global_mlp<<<dim3(GG), dim3(GOUTD), 0, stream>>>(agg, globals_, Wg, bg, out_g);
}
